// GraphSSDLayer_80934363725836
// MI455X (gfx1250) — matrix operand order repaired and checked by reading, not compiled, not run on hardware
//
#include <hip/hip_runtime.h>
#include <hip/hip_bf16.h>
#include <math.h>

#define BN 8
#define LG 1024
#define NH 8
#define HD 16
#define DS 64
#define NNODES (BN * LG)
#define TOLV 0.1f

typedef __attribute__((ext_vector_type(16))) _Float16 v16h_t;
typedef __attribute__((ext_vector_type(8)))  _Float16 v8h_t;
typedef __attribute__((ext_vector_type(8)))  float    v8f_t;

// ---------------------------------------------------------------------------
// WMMA helpers (CDNA5 16x16x32 f16 -> f32)
// ---------------------------------------------------------------------------
__device__ __forceinline__ v8f_t wmma16(v16h_t a, v16h_t b, v8f_t c) {
  return __builtin_amdgcn_wmma_f32_16x16x32_f16(false, a, false, b, (short)0, c,
                                                false, false);
}

__device__ __forceinline__ v8f_t zero8() {
  v8f_t z;
#pragma unroll
  for (int i = 0; i < 8; ++i) z[i] = 0.0f;
  return z;
}

// A operand: 16x32 tile, source row-major (ld elements per row).
// Lane (0..15): row M=lane, Ks {0..7, 16..23}; lane (16..31): row M=lane-16,
// Ks {8..15, 24..31}  (per CDNA5 ISA 16-bit A layout).
__device__ __forceinline__ v16h_t load_a16x32(const _Float16* __restrict__ A,
                                              int ld, int row0, int k0) {
  const int lane = threadIdx.x & 31;
  const int m = lane & 15;
  const int klo = (lane >> 4) << 3;  // 0 or 8
  const _Float16* p = A + (size_t)(row0 + m) * (size_t)ld + (size_t)(k0 + klo);
  v8h_t lo = *(const v8h_t*)p;          // K = klo..klo+7
  v8h_t hi = *(const v8h_t*)(p + 16);   // K = klo+16..klo+23
  return __builtin_shufflevector(lo, hi, 0, 1, 2, 3, 4, 5, 6, 7, 8, 9, 10, 11,
                                 12, 13, 14, 15);
}

// B operand: 32x16 tile, source stored column-major (column stride = ld).
// Lane (0..15): col N=lane, Ks {0..7, 16..23}; lane (16..31): col N=lane-16,
// Ks {8..15, 24..31}.
__device__ __forceinline__ v16h_t load_b32x16(const _Float16* __restrict__ Bcm,
                                              int ld, int k0, int col0) {
  const int lane = threadIdx.x & 31;
  const int n = lane & 15;
  const int klo = (lane >> 4) << 3;  // 0 or 8
  const _Float16* p =
      Bcm + (size_t)(col0 + n) * (size_t)ld + (size_t)(k0 + klo);
  v8h_t lo = *(const v8h_t*)p;        // K = klo..klo+7
  v8h_t hi = *(const v8h_t*)(p + 16);  // K = klo+16..klo+23
  return __builtin_shufflevector(lo, hi, 0, 1, 2, 3, 4, 5, 6, 7, 8, 9, 10, 11,
                                 12, 13, 14, 15);
}

// C/D layout: lane(0..15): col N=lane, rows M=0..7 in acc[0..7];
//             lane(16..31): col N=lane-16, rows M=8..15.
__device__ __forceinline__ void store_d_cm_f16(_Float16* __restrict__ D, int ld,
                                               int row0, int col0, v8f_t acc) {
  const int lane = threadIdx.x & 31;
  const int n = lane & 15;
  const int m0 = (lane >> 4) << 3;
  _Float16* p = D + (size_t)(col0 + n) * (size_t)ld + (size_t)(row0 + m0);
  v8h_t hv;
#pragma unroll
  for (int i = 0; i < 8; ++i) hv[i] = (_Float16)acc[i];
  *(v8h_t*)p = hv;  // 8 consecutive rows of one column: contiguous
}

// ---------------------------------------------------------------------------
// Stage A: per-node/head prep
// ---------------------------------------------------------------------------
__global__ __launch_bounds__(256) void k_prep(const float* __restrict__ dt,
                                              const float* __restrict__ dt_bias,
                                              float* __restrict__ ea0,
                                              float* __restrict__ ea1,
                                              float* __restrict__ dumm,
                                              _Float16* __restrict__ sdt16) {
  int idx = blockIdx.x * 256 + threadIdx.x;
  if (idx >= NNODES * NH) return;
  int n = idx >> 3, h = idx & 7;
  float bias = dt_bias[h];
  const float* p = dt + (size_t)n * (4 * NH) + 4 * h;
  auto sp = [](float x) { return x > 20.f ? x : log1pf(expf(x)); };
  ea0[idx] = expf(-0.5f * sp(p[0] + bias));
  ea1[idx] = expf(-0.5f * sp(p[1] + bias));
  dumm[idx] = expf(-sp(p[2] + bias));
  int b = n >> 10, l = n & 1023;
  sdt16[((size_t)(b * NH + h) << 10) + l] = (_Float16)p[3];
}

__global__ __launch_bounds__(256) void k_zero4(float4* __restrict__ p,
                                               size_t n4) {
  size_t i = (size_t)blockIdx.x * 256 + threadIdx.x;
  if (i < n4) p[i] = make_float4(0.f, 0.f, 0.f, 0.f);
}

__global__ __launch_bounds__(256) void k_scatter(const int* __restrict__ ei,
                                                 int E,
                                                 const float* __restrict__ ea0,
                                                 const float* __restrict__ ea1,
                                                 float* __restrict__ A32) {
  int idx = blockIdx.x * 256 + threadIdx.x;
  if (idx >= E * 8) return;
  int e = idx >> 3, h = idx & 7;
  int src = ei[e], dst = ei[E + e];
  int b = src >> 10, sl = src & 1023, dl = dst & 1023;
  float v = ea0[src * 8 + h] * ea1[dst * 8 + h];
  atomicAdd(A32 + (((size_t)(b * 8 + h)) << 20) + ((size_t)sl << 10) + dl, v);
}

// Row-normalize A[b,h,sl,:], write last = A_norm^T into rm+cm f16 layouts,
// and out0 = I + last.
__global__ __launch_bounds__(256) void k_normalize(
    const float* __restrict__ A32, const float* __restrict__ dumm,
    _Float16* __restrict__ Lrm, _Float16* __restrict__ Lcm,
    _Float16* __restrict__ Orm) {
  __shared__ float red[256];
  int i = blockIdx.x, bh = blockIdx.y;
  int b = bh >> 3, h = bh & 7;
  const float* row = A32 + ((size_t)bh << 20) + ((size_t)i << 10);
  float s = 0.f;
  for (int j = threadIdx.x; j < LG; j += 256) s += row[j];
  red[threadIdx.x] = s;
  __syncthreads();
  for (int st = 128; st > 0; st >>= 1) {
    if ((int)threadIdx.x < st) red[threadIdx.x] += red[threadIdx.x + st];
    __syncthreads();
  }
  float inv = 1.f / (red[0] + dumm[(size_t)(b * LG + i) * 8 + h] + TOLV);
  size_t base = (size_t)bh << 20;
  for (int j = threadIdx.x; j < LG; j += 256) {
    float v = row[j] * inv;
    _Float16 hv = (_Float16)v;
    // last[j][i] = A_norm[i][j]; Lcm indexed [col i][row j] == A_norm row-major
    Lcm[base + ((size_t)i << 10) + j] = hv;
    Lrm[base + ((size_t)j << 10) + i] = hv;
    Orm[base + ((size_t)j << 10) + i] = (_Float16)(v + (j == i ? 1.f : 0.f));
  }
}

__global__ __launch_bounds__(256) void k_cvt(const float* __restrict__ s,
                                             _Float16* __restrict__ d, int n) {
  int i = blockIdx.x * 256 + threadIdx.x;
  if (i < n) d[i] = (_Float16)s[i];
}

// xT[(b*8+h)*16 + hd][t] = x[b*1024+t, h*16+hd]
__global__ __launch_bounds__(256) void k_xT(const float* __restrict__ x,
                                            _Float16* __restrict__ xT) {
  int idx = blockIdx.x * 256 + threadIdx.x;
  if (idx >= BN * NH * HD * LG) return;
  int t = idx & 1023;
  int r = idx >> 10;
  int hd = r & 15;
  int bh = r >> 4;
  int b = bh >> 3, h = bh & 7;
  xT[idx] = (_Float16)x[(size_t)(b * LG + t) * 128 + h * 16 + hd];
}

// ---------------------------------------------------------------------------
// Batched WMMA GEMM: D = A(rm) @ B(cm) [+ Cin], f16 in, f32 accumulate.
// 256 threads = 8 waves arranged 4(M) x 2(N); wave tile 64x64 (16 WMMA/k-step).
// Block tile 256(M) x 128(N).
// Row-major result goes through a per-wave LDS staging buffer so global
// stores are fully-coalesced 128B runs; ADD_C folds "D = A@B + Cin" into the
// same coalesced pass (no strided C loads).
// ---------------------------------------------------------------------------
template <bool ADD_C, bool WRITE_CM>
__global__ __launch_bounds__(256) void k_gemm(
    const _Float16* __restrict__ A, int lda, size_t strideA,
    const _Float16* __restrict__ Bcm, int ldb, size_t strideB,
    const _Float16* __restrict__ Cin, _Float16* __restrict__ Drm,
    _Float16* __restrict__ Dcm, int ldd, size_t strideD, int K) {
  __shared__ _Float16 lbuf[8 * 32 * 64];  // 32 KB: per-wave 32x64 f16 stage
  const int w = threadIdx.x >> 5;
  const int lane = threadIdx.x & 31;
  const int wm = w >> 1, wn = w & 1;
  const int bm = blockIdx.y * 256 + wm * 64;
  const int bn = blockIdx.x * 128 + wn * 64;
  const _Float16* Ab = A + strideA * blockIdx.z;
  const _Float16* Bb = Bcm + strideB * blockIdx.z;

  v8f_t acc[4][4];
#pragma unroll
  for (int mi = 0; mi < 4; ++mi)
#pragma unroll
    for (int ni = 0; ni < 4; ++ni) acc[mi][ni] = zero8();

  for (int k0 = 0; k0 < K; k0 += 32) {
    v16h_t a[4];
#pragma unroll
    for (int mi = 0; mi < 4; ++mi)
      a[mi] = load_a16x32(Ab, lda, bm + mi * 16, k0);
#pragma unroll
    for (int ni = 0; ni < 4; ++ni) {
      v16h_t bt = load_b32x16(Bb, ldb, k0, bn + ni * 16);
#pragma unroll
      for (int mi = 0; mi < 4; ++mi)
        acc[mi][ni] = wmma16(a[mi], bt, acc[mi][ni]);
    }
  }

  // Column-major copy: contiguous v8h stores straight from the D layout.
  if (WRITE_CM) {
    _Float16* Dc = Dcm + strideD * blockIdx.z;
#pragma unroll
    for (int mi = 0; mi < 4; ++mi)
#pragma unroll
      for (int ni = 0; ni < 4; ++ni)
        store_d_cm_f16(Dc, ldd, bm + mi * 16, bn + ni * 16, acc[mi][ni]);
  }

  // Row-major result via per-wave LDS transpose (no inter-wave sharing, so
  // per-wave in-order LDS semantics make barriers unnecessary).
  _Float16* Dr = Drm + strideD * blockIdx.z;
  const _Float16* Cb = Cin + strideD * blockIdx.z;  // only read when ADD_C
  _Float16* wb = lbuf + w * (32 * 64);
  const int n = lane & 15;
  const int m0 = (lane >> 4) << 3;
  const int rsub = lane >> 3;  // 0..3
  const int chunk = lane & 7;  // 0..7
#pragma unroll
  for (int half = 0; half < 2; ++half) {
    // scatter f16 accumulators into LDS (row-major wave tile half: 32x64)
#pragma unroll
    for (int mi = 0; mi < 2; ++mi)
#pragma unroll
      for (int ni = 0; ni < 4; ++ni) {
        v8f_t a = acc[half * 2 + mi][ni];
        _Float16* q = wb + (mi * 16 + m0) * 64 + ni * 16 + n;
#pragma unroll
        for (int i = 0; i < 8; ++i) q[i * 64] = (_Float16)a[i];
      }
    // coalesced read-back + global store (8 lanes cover one 128B row run)
#pragma unroll
    for (int g = 0; g < 8; ++g) {
      int r = g * 4 + rsub;  // 0..31
      v8h_t v = *(const v8h_t*)(wb + r * 64 + chunk * 8);
      int grow = bm + half * 32 + r;
      _Float16* gp = Dr + (size_t)grow * (size_t)ldd + bn + chunk * 8;
      if (ADD_C) {
        v8h_t cv =
            *(const v8h_t*)(Cb + (size_t)grow * (size_t)ldd + bn + chunk * 8);
        v = v + cv;
      }
      *(v8h_t*)gp = v;
    }
  }
}

// ---------------------------------------------------------------------------
// Final: y[b,l, hd*8+h] = sum_t out[b,h,l,t]*G[b,l,t]*sdt[b,h,t]*x[b,t,h,hd]
//                         + x[b,l,h,hd]*D[h]
// Each wave: one 16-row M tile, N=16 (=HD), K=1024.
// ---------------------------------------------------------------------------
__global__ __launch_bounds__(256) void k_final(
    const _Float16* __restrict__ Orm, const _Float16* __restrict__ G16,
    const _Float16* __restrict__ sdt16, const _Float16* __restrict__ xT,
    const float* __restrict__ x, const float* __restrict__ Dvec,
    float* __restrict__ y) {
  const int w = threadIdx.x >> 5;
  const int lane = threadIdx.x & 31;
  const int bh = blockIdx.z;
  const int b = bh >> 3, h = bh & 7;
  const int row0 = (blockIdx.x * 8 + w) * 16;
  const int m = lane & 15;
  const int kloA = (lane >> 4) << 3;
  const int kloB = (lane >> 4) << 3;
  const _Float16* orow = Orm + ((size_t)bh << 20) + ((size_t)(row0 + m) << 10);
  const _Float16* grow = G16 + ((size_t)b << 20) + ((size_t)(row0 + m) << 10);
  const _Float16* srow = sdt16 + ((size_t)bh << 10);
  const _Float16* xcol = xT + ((size_t)(bh * 16 + m) << 10);
  v8f_t acc = zero8();
  for (int k0 = 0; k0 < LG; k0 += 32) {
    v8h_t olo = *(const v8h_t*)(orow + k0 + kloA);
    v8h_t ohi = *(const v8h_t*)(orow + k0 + kloA + 16);
    v8h_t glo = *(const v8h_t*)(grow + k0 + kloA);
    v8h_t ghi = *(const v8h_t*)(grow + k0 + kloA + 16);
    v8h_t slo = *(const v8h_t*)(srow + k0 + kloA);
    v8h_t shi = *(const v8h_t*)(srow + k0 + kloA + 16);
    v8h_t alo = olo * glo * slo;
    v8h_t ahi = ohi * ghi * shi;
    v16h_t av = __builtin_shufflevector(alo, ahi, 0, 1, 2, 3, 4, 5, 6, 7, 8, 9,
                                        10, 11, 12, 13, 14, 15);
    v8h_t xlo = *(const v8h_t*)(xcol + k0 + kloB);
    v8h_t xhi = *(const v8h_t*)(xcol + k0 + kloB + 16);
    v16h_t bv = __builtin_shufflevector(xlo, xhi, 0, 1, 2, 3, 4, 5, 6, 7, 8, 9,
                                        10, 11, 12, 13, 14, 15);
    acc = wmma16(av, bv, acc);
  }
  const int n = lane & 15;  // hd
  const int m0 = (lane >> 4) << 3;
  float Dh = Dvec[h];
#pragma unroll
  for (int i = 0; i < 8; ++i) {
    int l = row0 + m0 + i;
    size_t node = (size_t)(b * LG + l);
    // output channel = hd*NH + h (reference: transpose(0,2,3,1).reshape)
    y[node * 128 + (size_t)(n * 8 + h)] =
        acc[i] + x[node * 128 + (size_t)(h * 16 + n)] * Dh;
  }
}

// ---------------------------------------------------------------------------
// Host orchestration
// ---------------------------------------------------------------------------
extern "C" void kernel_launch(void* const* d_in, const int* in_sizes, int n_in,
                              void* d_out, int out_size, void* d_ws,
                              size_t ws_size, hipStream_t stream) {
  const float* x = (const float*)d_in[0];
  const float* B_mat = (const float*)d_in[1];
  const float* C_mat = (const float*)d_in[2];
  const float* dt = (const float*)d_in[3];
  const float* dt_bias = (const float*)d_in[4];
  const float* Dv = (const float*)d_in[5];
  const int* ei = (const int*)d_in[6];
  const int E = in_sizes[6] / 2;
  float* y = (float*)d_out;

  const size_t MATF32 = sizeof(float) * 64ull * LG * LG;     // 256 MB
  const size_t MATF16 = sizeof(_Float16) * 64ull * LG * LG;  // 128 MB

  char* p = (char*)d_ws;
  auto alloc = [&](size_t bytes) {
    void* r = (void*)p;
    p += (bytes + 255) & ~(size_t)255;
    return r;
  };
  float* A32 = (float*)alloc(MATF32);
  _Float16* L0rm = (_Float16*)alloc(MATF16);
  _Float16* L0cm = (_Float16*)alloc(MATF16);
  _Float16* Out0 = (_Float16*)alloc(MATF16);
  _Float16* Out1 = (_Float16*)alloc(MATF16);
  _Float16* G16 = (_Float16*)alloc(sizeof(_Float16) * (size_t)BN * LG * LG);
  _Float16* Cd16 = (_Float16*)alloc(sizeof(_Float16) * (size_t)NNODES * DS);
  _Float16* Bd16 = (_Float16*)alloc(sizeof(_Float16) * (size_t)NNODES * DS);
  _Float16* xT16 =
      (_Float16*)alloc(sizeof(_Float16) * (size_t)BN * NH * HD * LG);
  float* ea0 = (float*)alloc(sizeof(float) * (size_t)NNODES * NH);
  float* ea1 = (float*)alloc(sizeof(float) * (size_t)NNODES * NH);
  float* dumm = (float*)alloc(sizeof(float) * (size_t)NNODES * NH);
  _Float16* sdt16 = (_Float16*)alloc(sizeof(_Float16) * (size_t)BN * NH * LG);
  if ((size_t)(p - (char*)d_ws) > ws_size) return;  // workspace too small

  // A32 (fp32 scatter buffer) is dead after k_normalize: reuse as ping-pong L1.
  _Float16* L1rm = (_Float16*)A32;
  _Float16* L1cm = (_Float16*)A32 + 64ull * LG * LG;

  dim3 blk(256);
  const size_t MSTRIDE = (size_t)LG * LG;

  k_prep<<<(NNODES * NH + 255) / 256, blk, 0, stream>>>(dt, dt_bias, ea0, ea1,
                                                        dumm, sdt16);
  {
    size_t n4 = 64ull * LG * LG / 4;
    k_zero4<<<(unsigned)((n4 + 255) / 256), blk, 0, stream>>>((float4*)A32, n4);
  }
  k_scatter<<<(E * 8 + 255) / 256, blk, 0, stream>>>(ei, E, ea0, ea1, A32);
  k_normalize<<<dim3(LG, 64), blk, 0, stream>>>(A32, dumm, L0rm, L0cm, Out0);

  k_cvt<<<(NNODES * DS + 255) / 256, blk, 0, stream>>>(C_mat, Cd16,
                                                       NNODES * DS);
  k_cvt<<<(NNODES * DS + 255) / 256, blk, 0, stream>>>(B_mat, Bd16,
                                                       NNODES * DS);
  k_xT<<<(BN * NH * HD * LG + 255) / 256, blk, 0, stream>>>(x, xT16);

  // G[b,l,t] = sum_d C[l,d]*B[t,d]; A=Cd16 rm (ld=64), Bop = Bd16 rows (=cm).
  k_gemm<false, false><<<dim3(8, 4, BN), blk, 0, stream>>>(
      Cd16, DS, (size_t)LG * DS, Bd16, DS, (size_t)LG * DS, nullptr, G16,
      nullptr, LG, MSTRIDE, DS);

  // Doubling chain: 5 iterations of { last = last@last; out = out + out@last }
  _Float16 *Lsr = L0rm, *Lsc = L0cm, *Ldr = L1rm, *Ldc = L1cm;
  _Float16 *Osrc = Out0, *Odst = Out1;
  for (int it = 0; it < 5; ++it) {
    k_gemm<false, true><<<dim3(8, 4, 64), blk, 0, stream>>>(
        Lsr, LG, MSTRIDE, Lsc, LG, MSTRIDE, nullptr, Ldr, Ldc, LG, MSTRIDE, LG);
    k_gemm<true, false><<<dim3(8, 4, 64), blk, 0, stream>>>(
        Osrc, LG, MSTRIDE, Ldc, LG, MSTRIDE, Osrc, Odst, nullptr, LG, MSTRIDE,
        LG);
    _Float16* t;
    t = Lsr; Lsr = Ldr; Ldr = t;
    t = Lsc; Lsc = Ldc; Ldc = t;
    t = Osrc; Osrc = Odst; Odst = t;
  }

  k_final<<<dim3(8, 1, 64), blk, 0, stream>>>(Osrc, G16, sdt16, xT16, x, Dv, y);
}
